// MultiHeadAttention_47270410060490
// MI455X (gfx1250) — compile-verified, matrix operand order checked by reading, not run on hardware
//
#include <hip/hip_runtime.h>
#include <hip/hip_bf16.h>

// ---------------------------------------------------------------------------
// MHA forward for MI455X (gfx1250), wave32.
//   - all matmuls on v_wmma_f32_16x16x32_f16
//   - LDS staging via global_load_async_to_lds_b128 (ASYNCcnt), double-buffered
//     K-panels in the projection GEMMs
//   - attn (536 MB, write-once) stored with nontemporal float4 stores
// d_out = [ out : 2*2048*1024 f32 | attn : 2*16*2048*2048 f32 ]
// ---------------------------------------------------------------------------

typedef _Float16 f16;
typedef __attribute__((ext_vector_type(16))) _Float16 v16h;
typedef __attribute__((ext_vector_type(8)))  float    v8f;
typedef __attribute__((ext_vector_type(4)))  float    v4f;

#define WMMA_F16(a, b, c) \
    __builtin_amdgcn_wmma_f32_16x16x32_f16(false, (a), false, (b), (short)0, (c), false, false)

// Async DMA: 16 bytes global -> LDS per lane, tracked by ASYNCcnt.
__device__ __forceinline__ void async_copy_b128(const f16* g, f16* l) {
    // generic LDS pointer: low 32 bits are the LDS byte address (ISA 10.2 aperture rules)
    unsigned loff = (unsigned)(unsigned long long)l;
    asm volatile("global_load_async_to_lds_b128 %0, %1, off"
                 :: "v"(loff), "v"((unsigned long long)g)
                 : "memory");
}

__device__ __forceinline__ void wait_async0() {
#if __has_builtin(__builtin_amdgcn_s_wait_asynccnt)
    __builtin_amdgcn_s_wait_asynccnt(0);
#else
    asm volatile("s_wait_asynccnt 0x0" ::: "memory");
#endif
}

// 16x32 (MxK) 16-bit A/B fragment from a row-major tile (ISA 7.12.2 layout):
// lane = M (or N), eight 32-bit K-pairs per lane with the lane-half K split.
__device__ __forceinline__ v16h frag_load(const f16* tile, int stride, int lane) {
    union { v16h v; unsigned u[8]; } r;
    const int m  = lane & 15;
    const int hi = lane >> 4;
    const f16* row = tile + m * stride;
#pragma unroll
    for (int i = 0; i < 8; ++i) {
        const int k = ((i & 4) << 2) + (hi << 3) + ((i & 3) << 1); // even
        r.u[i] = *(const unsigned*)(row + k);
    }
    return r.v;
}

// ---------------------------------------------------------------------------
__global__ __launch_bounds__(256) void k_f32_to_f16(const float* __restrict__ in,
                                                    f16* __restrict__ out, int n) {
    int i = blockIdx.x * 256 + threadIdx.x;
    if (i < n) out[i] = (f16)in[i];
}

// ---------------------------------------------------------------------------
// QKV projection: [4096 x 1024] x W^T, three weights via blockIdx.z.
// Double-buffered async K-panels; output f16 in [b,h,s,d] layout.
// ---------------------------------------------------------------------------
__global__ __launch_bounds__(256) void k_qkv_gemm(const f16* __restrict__ H,
                                                  const f16* __restrict__ Wq,
                                                  const f16* __restrict__ Wk,
                                                  const f16* __restrict__ Wv,
                                                  f16* __restrict__ Qo,
                                                  f16* __restrict__ Ko,
                                                  f16* __restrict__ Vo) {
    __shared__ f16 As[2][64][64];
    __shared__ f16 Bs[2][64][64];

    const int z  = blockIdx.z;
    const f16* W = (z == 0) ? Wq : (z == 1) ? Wk : Wv;
    f16* dst     = (z == 0) ? Qo : (z == 1) ? Ko : Vo;

    const int m0   = blockIdx.y * 64;
    const int n0   = blockIdx.x * 64;
    const int tid  = threadIdx.x;
    const int lane = tid & 31;
    const int wv   = tid >> 5;
    const int tm   = wv >> 1;
    const int tn0  = (wv & 1) * 2;
    const int cr   = tid >> 2;
    const int cc   = (tid & 3) * 16;

    const f16* arow = H + (size_t)(m0 + cr) * 1024 + cc;
    const f16* brow = W + (size_t)(n0 + cr) * 1024 + cc;

    v8f c0 = {0.f,0.f,0.f,0.f,0.f,0.f,0.f,0.f};
    v8f c1 = c0;

    // prologue: stage panel 0 into buffer 0
    async_copy_b128(arow,     &As[0][cr][cc]);
    async_copy_b128(arow + 8, &As[0][cr][cc + 8]);
    async_copy_b128(brow,     &Bs[0][cr][cc]);
    async_copy_b128(brow + 8, &Bs[0][cr][cc + 8]);

    int cur = 0;
    for (int kk = 0; kk < 1024; kk += 64, cur ^= 1) {
        wait_async0();
        __syncthreads();
        if (kk + 64 < 1024) {   // async-prefetch next panel while computing
            const int nxt = cur ^ 1, ko = kk + 64;
            async_copy_b128(arow + ko,     &As[nxt][cr][cc]);
            async_copy_b128(arow + ko + 8, &As[nxt][cr][cc + 8]);
            async_copy_b128(brow + ko,     &Bs[nxt][cr][cc]);
            async_copy_b128(brow + ko + 8, &Bs[nxt][cr][cc + 8]);
        }

        v16h a0 = frag_load(&As[cur][tm * 16][0], 64, lane);
        v16h a1 = frag_load(&As[cur][tm * 16][32], 64, lane);
        v16h b0 = frag_load(&Bs[cur][tn0 * 16][0], 64, lane);
        v16h b1 = frag_load(&Bs[cur][tn0 * 16][32], 64, lane);
        c0 = WMMA_F16(a0, b0, c0);
        c0 = WMMA_F16(a1, b1, c0);
        v16h b2 = frag_load(&Bs[cur][(tn0 + 1) * 16][0], 64, lane);
        v16h b3 = frag_load(&Bs[cur][(tn0 + 1) * 16][32], 64, lane);
        c1 = WMMA_F16(a0, b2, c1);
        c1 = WMMA_F16(a1, b3, c1);
        __syncthreads();
    }

    const int nlo = lane & 15;
    const int mhi = (lane >> 4) * 8;
#pragma unroll
    for (int t = 0; t < 2; ++t) {
        const v8f c = t ? c1 : c0;
        const int ng = n0 + (tn0 + t) * 16 + nlo;
        const int h  = ng >> 6, d = ng & 63;
#pragma unroll
        for (int r = 0; r < 8; ++r) {
            const int mg = m0 + tm * 16 + mhi + r;
            const int b  = mg >> 11, s = mg & 2047;
            dst[(((size_t)b * 16 + h) * 2048 + s) * 64 + d] = (f16)c[r];
        }
    }
}

// ---------------------------------------------------------------------------
// RoPE in place on Q and K (f16, [bh,s,64]), rotate-half pairs d,d+32.
// ---------------------------------------------------------------------------
__global__ __launch_bounds__(256) void k_rope(f16* __restrict__ Q, f16* __restrict__ K) {
    const int tid = blockIdx.x * 256 + threadIdx.x;   // 2*16*2048*32 pairs
    const int dp  = tid & 31;
    const int s   = (tid >> 5) & 2047;
    const int bh  = tid >> 16;
    const size_t base = ((size_t)bh * 2048 + s) * 64;

    const float inv = __powf(10000.f, -((float)(2 * dp)) * (1.f / 64.f));
    const float ang = (float)s * inv;
    const float cs = __cosf(ang), sn = __sinf(ang);

    float x1 = (float)Q[base + dp], x2 = (float)Q[base + dp + 32];
    Q[base + dp]      = (f16)(x1 * cs - x2 * sn);
    Q[base + dp + 32] = (f16)(x1 * sn + x2 * cs);

    x1 = (float)K[base + dp]; x2 = (float)K[base + dp + 32];
    K[base + dp]      = (f16)(x1 * cs - x2 * sn);
    K[base + dp + 32] = (f16)(x1 * sn + x2 * cs);
}

// ---------------------------------------------------------------------------
// Fused causal attention: one workgroup = (b*h, 64-query block).
// Pass 1: online row max/sum (parallel partials). Pass 2: recompute S, write
// normalized attn once (coalesced NT float4), accumulate O = P V in VGPRs.
// ---------------------------------------------------------------------------
__global__ __launch_bounds__(256) void k_attn(const f16* __restrict__ Q,
                                              const f16* __restrict__ K,
                                              const f16* __restrict__ V,
                                              f16* __restrict__ O,
                                              float* __restrict__ attn) {
    __shared__ f16   Qs[64][64];
    __shared__ f16   Ks[64][64];
    __shared__ f16   Vts[64][72];   // V transposed: [dim][key]
    __shared__ f16   Ps[64][72];
    __shared__ float Ss[64][68];
    __shared__ float pm[64][4];
    __shared__ float ps[64][4];
    __shared__ float rmax[64];
    __shared__ float rsum[64];

    const int qb = blockIdx.x;        // query block (0..31)
    const int bh = blockIdx.y;        // b*16+h
    const f16* Qh = Q + (size_t)bh * 2048 * 64;
    const f16* Kh = K + (size_t)bh * 2048 * 64;
    const f16* Vh = V + (size_t)bh * 2048 * 64;
    float* attnh  = attn + (size_t)bh * 2048 * 2048;

    const int tid  = threadIdx.x;
    const int lane = tid & 31;
    const int wv   = tid >> 5;
    const int tm   = wv >> 1;
    const int tn0  = (wv & 1) * 2;
    const int nlo  = lane & 15;
    const int mhi  = (lane >> 4) * 8;
    const int cr   = tid >> 2;
    const int cc   = (tid & 3) * 16;
    const float scale = 0.125f;       // 1/sqrt(64)

    // async-stage Q block
    async_copy_b128(Qh + (size_t)(qb * 64 + cr) * 64 + cc,     &Qs[cr][cc]);
    async_copy_b128(Qh + (size_t)(qb * 64 + cr) * 64 + cc + 8, &Qs[cr][cc + 8]);
    if (tid < 64) { rmax[tid] = -3.0e38f; rsum[tid] = 0.f; }
    wait_async0();
    __syncthreads();

    // ---- pass 1: row statistics ----
    for (int j = 0; j <= qb; ++j) {
        async_copy_b128(Kh + (size_t)(j * 64 + cr) * 64 + cc,     &Ks[cr][cc]);
        async_copy_b128(Kh + (size_t)(j * 64 + cr) * 64 + cc + 8, &Ks[cr][cc + 8]);
        wait_async0();
        __syncthreads();

        v16h a0 = frag_load(&Qs[tm * 16][0], 64, lane);
        v16h a1 = frag_load(&Qs[tm * 16][32], 64, lane);
#pragma unroll
        for (int t = 0; t < 2; ++t) {
            const int tn = tn0 + t;
            v16h b0 = frag_load(&Ks[tn * 16][0], 64, lane);
            v16h b1 = frag_load(&Ks[tn * 16][32], 64, lane);
            v8f c = {0.f,0.f,0.f,0.f,0.f,0.f,0.f,0.f};
            c = WMMA_F16(a0, b0, c);
            c = WMMA_F16(a1, b1, c);
            const int col = tn * 16 + nlo;
#pragma unroll
            for (int r = 0; r < 8; ++r) {
                const int row = tm * 16 + mhi + r;
                float sval = c[r] * scale;
                if (j == qb && col > row) sval = -__builtin_inff();
                Ss[row][col] = sval;
            }
        }
        __syncthreads();

        { // 256-thread partial online max/sum (4 partials per row)
            const int row = tid & 63, q = tid >> 6;
            float m = -3.0e38f, su = 0.f;
            const float* srow = &Ss[row][q * 16];
#pragma unroll
            for (int c = 0; c < 16; ++c) {
                const float v = srow[c];
                if (v > m) { su = su * __expf(m - v) + 1.f; m = v; }
                else       { su += __expf(v - m); }
            }
            pm[row][q] = m; ps[row][q] = su;
        }
        __syncthreads();

        if (tid < 64) { // merge partials with running stats
            float m = rmax[tid], su = rsum[tid];
#pragma unroll
            for (int q = 0; q < 4; ++q) {
                const float pq = pm[tid][q];
                const float mn = fmaxf(m, pq);
                su = su * __expf(m - mn) + ps[tid][q] * __expf(pq - mn);
                m = mn;
            }
            rmax[tid] = m; rsum[tid] = su;
        }
        __syncthreads();
    }

    // ---- pass 2: normalized attn store + O accumulation ----
    v8f o0 = {0.f,0.f,0.f,0.f,0.f,0.f,0.f,0.f};
    v8f o1 = o0;

    for (int j = 0; j <= qb; ++j) {
        async_copy_b128(Kh + (size_t)(j * 64 + cr) * 64 + cc,     &Ks[cr][cc]);
        async_copy_b128(Kh + (size_t)(j * 64 + cr) * 64 + cc + 8, &Ks[cr][cc + 8]);
        { // V: load row, scatter-transpose into Vts[dim][key]
            union { uint4 q[2]; f16 h[16]; } tv;
            const uint4* vs = (const uint4*)(Vh + (size_t)(j * 64 + cr) * 64 + cc);
            tv.q[0] = vs[0]; tv.q[1] = vs[1];
#pragma unroll
            for (int i = 0; i < 16; ++i) Vts[cc + i][cr] = tv.h[i];
        }
        wait_async0();
        __syncthreads();

        v16h a0 = frag_load(&Qs[tm * 16][0], 64, lane);
        v16h a1 = frag_load(&Qs[tm * 16][32], 64, lane);
#pragma unroll
        for (int t = 0; t < 2; ++t) {
            const int tn = tn0 + t;
            v16h b0 = frag_load(&Ks[tn * 16][0], 64, lane);
            v16h b1 = frag_load(&Ks[tn * 16][32], 64, lane);
            v8f c = {0.f,0.f,0.f,0.f,0.f,0.f,0.f,0.f};
            c = WMMA_F16(a0, b0, c);
            c = WMMA_F16(a1, b1, c);
            const int col = tn * 16 + nlo;
#pragma unroll
            for (int r = 0; r < 8; ++r) {
                const int row = tm * 16 + mhi + r;
                float sval = c[r] * scale;
                if (j == qb && col > row) sval = -__builtin_inff();
                const float p = __expf(sval - rmax[row]) / rsum[row];
                Ss[row][col] = p;          // fp32 staging for coalesced store
                Ps[row][col] = (f16)p;     // f16 operand for P @ V
            }
        }
        __syncthreads();

        { // coalesced nontemporal attn store (write-once stream, bypass L2 retention)
            v4f* dstp = (v4f*)&attnh[(size_t)(qb * 64 + cr) * 2048 + j * 64 + cc];
            const float* srow = &Ss[cr][cc];
#pragma unroll
            for (int i = 0; i < 4; ++i)
                __builtin_nontemporal_store(*(const v4f*)(srow + 4 * i), dstp + i);
        }

        v16h pa0 = frag_load(&Ps[tm * 16][0], 72, lane);
        v16h pa1 = frag_load(&Ps[tm * 16][32], 72, lane);
        v16h vb0 = frag_load(&Vts[tn0 * 16][0], 72, lane);
        v16h vb1 = frag_load(&Vts[tn0 * 16][32], 72, lane);
        o0 = WMMA_F16(pa0, vb0, o0);
        o0 = WMMA_F16(pa1, vb1, o0);
        v16h vb2 = frag_load(&Vts[(tn0 + 1) * 16][0], 72, lane);
        v16h vb3 = frag_load(&Vts[(tn0 + 1) * 16][32], 72, lane);
        o1 = WMMA_F16(pa0, vb2, o1);
        o1 = WMMA_F16(pa1, vb3, o1);
        __syncthreads();
    }

    // zero-fill masked (upper-triangular) attn blocks, nontemporal
    {
        const v4f z = {0.f, 0.f, 0.f, 0.f};
        for (int j = qb + 1; j < 32; ++j) {
            v4f* d = (v4f*)&attnh[(size_t)(qb * 64 + cr) * 2048 + j * 64 + cc];
#pragma unroll
            for (int i = 0; i < 4; ++i) __builtin_nontemporal_store(z, d + i);
        }
    }

    // write O in [b, s, h*64+d] f16 layout for the output projection
    const int b = bh >> 4, h = bh & 15;
#pragma unroll
    for (int t = 0; t < 2; ++t) {
        const v8f o = t ? o1 : o0;
        const int d = (tn0 + t) * 16 + nlo;
#pragma unroll
        for (int r = 0; r < 8; ++r) {
            const int s = qb * 64 + tm * 16 + mhi + r;
            O[((size_t)b * 2048 + s) * 1024 + h * 64 + d] = (f16)o[r];
        }
    }
}

// ---------------------------------------------------------------------------
// Output projection: out = O @ Wo^T, fp32 store. Double-buffered async panels.
// ---------------------------------------------------------------------------
__global__ __launch_bounds__(256) void k_out_gemm(const f16* __restrict__ A,
                                                  const f16* __restrict__ W,
                                                  float* __restrict__ out) {
    __shared__ f16 As[2][64][64];
    __shared__ f16 Bs[2][64][64];

    const int m0   = blockIdx.y * 64;
    const int n0   = blockIdx.x * 64;
    const int tid  = threadIdx.x;
    const int lane = tid & 31;
    const int wv   = tid >> 5;
    const int tm   = wv >> 1;
    const int tn0  = (wv & 1) * 2;
    const int cr   = tid >> 2;
    const int cc   = (tid & 3) * 16;

    const f16* arow = A + (size_t)(m0 + cr) * 1024 + cc;
    const f16* brow = W + (size_t)(n0 + cr) * 1024 + cc;

    v8f c0 = {0.f,0.f,0.f,0.f,0.f,0.f,0.f,0.f};
    v8f c1 = c0;

    async_copy_b128(arow,     &As[0][cr][cc]);
    async_copy_b128(arow + 8, &As[0][cr][cc + 8]);
    async_copy_b128(brow,     &Bs[0][cr][cc]);
    async_copy_b128(brow + 8, &Bs[0][cr][cc + 8]);

    int cur = 0;
    for (int kk = 0; kk < 1024; kk += 64, cur ^= 1) {
        wait_async0();
        __syncthreads();
        if (kk + 64 < 1024) {
            const int nxt = cur ^ 1, ko = kk + 64;
            async_copy_b128(arow + ko,     &As[nxt][cr][cc]);
            async_copy_b128(arow + ko + 8, &As[nxt][cr][cc + 8]);
            async_copy_b128(brow + ko,     &Bs[nxt][cr][cc]);
            async_copy_b128(brow + ko + 8, &Bs[nxt][cr][cc + 8]);
        }

        v16h a0 = frag_load(&As[cur][tm * 16][0], 64, lane);
        v16h a1 = frag_load(&As[cur][tm * 16][32], 64, lane);
        v16h b0 = frag_load(&Bs[cur][tn0 * 16][0], 64, lane);
        v16h b1 = frag_load(&Bs[cur][tn0 * 16][32], 64, lane);
        c0 = WMMA_F16(a0, b0, c0);
        c0 = WMMA_F16(a1, b1, c0);
        v16h b2 = frag_load(&Bs[cur][(tn0 + 1) * 16][0], 64, lane);
        v16h b3 = frag_load(&Bs[cur][(tn0 + 1) * 16][32], 64, lane);
        c1 = WMMA_F16(a0, b2, c1);
        c1 = WMMA_F16(a1, b3, c1);
        __syncthreads();
    }

    const int nlo = lane & 15;
    const int mhi = (lane >> 4) * 8;
#pragma unroll
    for (int t = 0; t < 2; ++t) {
        const v8f c = t ? c1 : c0;
        const int ng = n0 + (tn0 + t) * 16 + nlo;
#pragma unroll
        for (int r = 0; r < 8; ++r) {
            const int mg = m0 + tm * 16 + mhi + r;
            out[(size_t)mg * 1024 + ng] = c[r];
        }
    }
}

// ---------------------------------------------------------------------------
// Host-side launch
// ---------------------------------------------------------------------------
extern "C" void kernel_launch(void* const* d_in, const int* in_sizes, int n_in,
                              void* d_out, int out_size, void* d_ws, size_t ws_size,
                              hipStream_t stream) {
    (void)in_sizes; (void)n_in; (void)out_size; (void)ws_size;

    const float* hs = (const float*)d_in[0];
    const float* Wq = (const float*)d_in[1];
    const float* Wk = (const float*)d_in[2];
    const float* Wv = (const float*)d_in[3];
    const float* Wo = (const float*)d_in[4];

    const size_t MB = 1024 * 1024;
    char* ws = (char*)d_ws;
    f16* Hf   = (f16*)(ws + 0);        //  8 MB  [4096 x 1024]
    f16* Wqf  = (f16*)(ws + 8  * MB);  //  2 MB each
    f16* Wkf  = (f16*)(ws + 10 * MB);
    f16* Wvf  = (f16*)(ws + 12 * MB);
    f16* Wof  = (f16*)(ws + 14 * MB);
    f16* Qf   = (f16*)(ws + 16 * MB);  //  8 MB  [bh, s, d]
    f16* Kf   = (f16*)(ws + 24 * MB);
    f16* Vf   = (f16*)(ws + 32 * MB);
    f16* Of   = (f16*)(ws + 40 * MB);  //  8 MB  [b, s, 1024]

    float* out_p  = (float*)d_out;
    float* attn_p = out_p + (size_t)2 * 2048 * 1024;

    k_f32_to_f16<<<16384, 256, 0, stream>>>(hs, Hf, 4096 * 1024);
    k_f32_to_f16<<<4096,  256, 0, stream>>>(Wq, Wqf, 1024 * 1024);
    k_f32_to_f16<<<4096,  256, 0, stream>>>(Wk, Wkf, 1024 * 1024);
    k_f32_to_f16<<<4096,  256, 0, stream>>>(Wv, Wvf, 1024 * 1024);
    k_f32_to_f16<<<4096,  256, 0, stream>>>(Wo, Wof, 1024 * 1024);

    k_qkv_gemm<<<dim3(16, 64, 3), 256, 0, stream>>>(Hf, Wqf, Wkf, Wvf, Qf, Kf, Vf);
    k_rope<<<8192, 256, 0, stream>>>(Qf, Kf);
    k_attn<<<dim3(32, 32), 256, 0, stream>>>(Qf, Kf, Vf, Of, attn_p);
    k_out_gemm<<<dim3(16, 64), 256, 0, stream>>>(Of, Wof, out_p);
}
